// SEM_26938034880814
// MI455X (gfx1250) — compile-verified, matrix operand order checked by reading, not run on hardware
//
#include <hip/hip_runtime.h>

// SEM forward for MI455X (gfx1250, wave32). Heavy matmuls (mat-GRU W@z / Uh@hcap,
// attention node_g@word^T and attn@attn^T) run through v_wmma_f32_16x16x32_bf16.
// GEMM v2: K-tile = 64 (2 WMMAs per barrier), n-major B tile in LDS so every
// fragment is two contiguous 16B runs -> ds_load_b128 assembly, uint4 global
// staging with hoisted base pointers, 72-element padded LDS rows (conflict-free
// b128 fragment reads), global_prefetch on the next A tile.

#define Bc 32
#define Nc 1000
#define Dc 512
#define Lc 128
#define Kc 256

typedef __attribute__((ext_vector_type(16))) __bf16 v16bf;
typedef __attribute__((ext_vector_type(8)))  float  v8f;

// ---------------- reductions ----------------
__device__ __forceinline__ float wave_sum(float v) {
#pragma unroll
  for (int o = 16; o > 0; o >>= 1) v += __shfl_down(v, o, 32);
  return v;
}

__device__ __forceinline__ float blk_sum(float v, float* red) {
  int t = threadIdx.x;
  red[t] = v; __syncthreads();
  for (int s = blockDim.x >> 1; s > 0; s >>= 1) {
    if (t < s) red[t] += red[t + s];
    __syncthreads();
  }
  float r = red[0]; __syncthreads();
  return r;
}

__device__ __forceinline__ float blk_max(float v, float* red) {
  int t = threadIdx.x;
  red[t] = v; __syncthreads();
  for (int s = blockDim.x >> 1; s > 0; s >>= 1) {
    if (t < s) red[t] = fmaxf(red[t], red[t + s]);
    __syncthreads();
  }
  float r = red[0]; __syncthreads();
  return r;
}

// ---------------- small dense helpers ----------------
// y[b,d] = tanh(x[b,:] @ W[:,d] + bias[d]),  x:[B,Din], W:[Din,Dout]
__global__ void linear_tanh_kernel(const float* __restrict__ x, const float* __restrict__ W,
                                   const float* __restrict__ bias, float* __restrict__ y,
                                   int Din, int Dout) {
  int i = blockIdx.x * blockDim.x + threadIdx.x;
  int b = i / Dout, d = i % Dout;
  if (b >= Bc) return;
  float acc = bias[d];
  const float* xb = x + (size_t)b * Din;
  for (int k = 0; k < Din; ++k) acc += xb[k] * W[(size_t)k * Dout + d];
  y[i] = tanhf(acc);
}

__global__ void rownorm_kernel(const float* __restrict__ v, float* __restrict__ norm) {
  __shared__ float red[256];
  int b = blockIdx.x;
  float s = 0.f;
  for (int d = threadIdx.x; d < Dc; d += blockDim.x) {
    float x = v[(size_t)b * Dc + d];
    s += x * x;
  }
  s = blk_sum(s, red);
  if (threadIdx.x == 0) norm[b] = sqrtf(s);
}

// scores[b,n] = (node_embs[b,n,:] . scorer[b,:]) / norm[b] ; one wave per row
__global__ void scores_kernel(const float* __restrict__ ne, const float* __restrict__ scorer,
                              const float* __restrict__ norm, float* __restrict__ scores) {
  int gw = (blockIdx.x * blockDim.x + threadIdx.x) >> 5;
  int lane = threadIdx.x & 31;
  if (gw >= Bc * Nc) return;
  int b = gw / Nc, n = gw % Nc;
  const float* row = ne + ((size_t)b * Nc + n) * Dc;
  const float* sc = scorer + (size_t)b * Dc;
  float acc = 0.f;
  for (int d = lane; d < Dc; d += 32) acc += row[d] * sc[d];
  acc = wave_sum(acc);
  if (lane == 0) scores[gw] = acc / norm[b];
}

// ---------------- top-k via bitonic sort (descending) ----------------
__global__ void topk_kernel(const float* __restrict__ scores, int* __restrict__ idx) {
  __shared__ float sv[1024];
  __shared__ int   si[1024];
  int b = blockIdx.x, t = threadIdx.x;           // 512 threads
  for (int i = t; i < 1024; i += 512) {
    sv[i] = (i < Nc) ? scores[(size_t)b * Nc + i] : -3.0e38f;
    si[i] = i;
  }
  __syncthreads();
  for (int k = 2; k <= 1024; k <<= 1) {
    for (int j = k >> 1; j > 0; j >>= 1) {
#pragma unroll
      for (int l = 0; l < 2; ++l) {
        int i = t + l * 512;
        int ixj = i ^ j;
        if (ixj > i) {
          bool desc = ((i & k) == 0);
          float a = sv[i], c = sv[ixj];
          bool sw = desc ? (a < c) : (a > c);
          if (sw) {
            sv[i] = c; sv[ixj] = a;
            int ta = si[i]; si[i] = si[ixj]; si[ixj] = ta;
          }
        }
      }
      __syncthreads();
    }
  }
  if (t < Kc) idx[(size_t)b * Kc + t] = si[t];
}

// entropy[b] = logZ - T/Z ; policy[b] = mean_k(logp[idx_k])
__global__ void stats_kernel(const float* __restrict__ scores, const int* __restrict__ idx,
                             float* __restrict__ policy, float* __restrict__ entropy) {
  __shared__ float red[256];
  int b = blockIdx.x, t = threadIdx.x;
  const float* s = scores + (size_t)b * Nc;
  float m = -3.0e38f;
  for (int i = t; i < Nc; i += 256) m = fmaxf(m, s[i]);
  m = blk_max(m, red);
  float z = 0.f, tt = 0.f;
  for (int i = t; i < Nc; i += 256) {
    float e = expf(s[i] - m);
    z += e; tt += e * (s[i] - m);
  }
  z  = blk_sum(z,  red);
  tt = blk_sum(tt, red);
  float logZ = logf(z);
  float ps = 0.f;
  for (int k = t; k < Kc; k += 256) ps += s[idx[(size_t)b * Kc + k]] - m - logZ;
  ps = blk_sum(ps, red);
  if (t == 0) {
    entropy[b] = logZ - tt / z;
    policy[b]  = ps / (float)Kc;
  }
}

// ---------------- gathers / conversions ----------------
// z[b,r,j] = node_embs[b, idx[b,j], r] * tanh(scores[b, idx[b,j]]), r<Kc (bf16)
__global__ void z_gather_kernel(const float* __restrict__ ne, const float* __restrict__ scores,
                                const int* __restrict__ idx, __bf16* __restrict__ z) {
  long i = (long)blockIdx.x * blockDim.x + threadIdx.x;
  if (i >= (long)Bc * Kc * Kc) return;
  int b = (int)(i / (Kc * Kc));
  int r = (int)((i / Kc) % Kc);
  int j = (int)(i % Kc);
  int node = idx[(size_t)b * Kc + j];
  float sval = scores[(size_t)b * Nc + node];
  float v = ne[((size_t)b * Nc + node) * Dc + r] * tanhf(sval);
  z[i] = (__bf16)v;
}

__global__ void nodeg_gather_kernel(const float* __restrict__ ne, const int* __restrict__ idx,
                                    __bf16* __restrict__ ng) {
  long i = (long)blockIdx.x * blockDim.x + threadIdx.x;
  if (i >= (long)Bc * Kc * Dc) return;
  int b = (int)(i / (Kc * Dc));
  int k = (int)((i / Dc) % Kc);
  int d = (int)(i % Dc);
  int node = idx[(size_t)b * Kc + k];
  ng[i] = (__bf16)ne[((size_t)b * Nc + node) * Dc + d];
}

__global__ void cvt_bf16_kernel(const float* __restrict__ s, __bf16* __restrict__ d, long n) {
  long i = (long)blockIdx.x * blockDim.x + threadIdx.x;
  if (i < n) d[i] = (__bf16)s[i];
}

// C[KxK] = A[KxK] @ B[KxK] + bias[KxK]  (one-off, shared across batch)
__global__ void mmK_bias_kernel(const float* __restrict__ A, const float* __restrict__ Bm,
                                const float* __restrict__ bias, float* __restrict__ C) {
  int i = blockIdx.x * blockDim.x + threadIdx.x;
  if (i >= Kc * Kc) return;
  int r = i / Kc, c = i % Kc;
  float acc = bias[i];
  for (int k = 0; k < Kc; ++k) acc += A[r * Kc + k] * Bm[k * Kc + c];
  C[i] = acc;
}

// ---------------- batched bf16 WMMA GEMM (v2) ----------------
// C[b](MxN,f32) (+)= A[b?](MxKd,bf16,row-major) x B[b?](KdxN, or NxKd if bTrans)
// grid = (N/16, M/64, batch), block = 128 (4 waves, one 16x16 C tile per wave).
// K-tile = 64 -> two v_wmma per barrier interval. LDS rows padded to 72 bf16
// (144B, 16B-aligned, conflict-free for b128 fragment reads). B tile stored
// n-major so both A and B fragments are two contiguous 16-byte runs.
#define KT 64
#define LDP 72
union Frag16 { v16bf v; uint4 q[2]; };

__global__ __launch_bounds__(128) void gemm_bf16_wmma(
    const __bf16* __restrict__ A, long strideA, int lda,
    const __bf16* __restrict__ Bm, long strideB, int ldb, int bTrans,
    float* __restrict__ C, long strideC, int ldc,
    int Kd, int accumulate) {
  __shared__ __bf16 As[64][LDP];   // [row][k]
  __shared__ __bf16 Bs[16][LDP];   // [n][k]  (n-major!)
  const int tid  = threadIdx.x;
  const int w    = tid >> 5, lane = tid & 31;
  const int half = lane >> 4, ln = lane & 15;
  const int y_base = blockIdx.y * 64;
  const int x0     = blockIdx.x * 16;
  const __bf16* Ab = A  + (size_t)blockIdx.z * strideA;
  const __bf16* Bb = Bm + (size_t)blockIdx.z * strideB;
  float*        Cb = C  + (size_t)blockIdx.z * strideC;

  // hoisted per-thread staging bases (A: 512 16B-chunks, 4 per thread)
  const __bf16* pA[4];
  int rA[4], cA[4];
#pragma unroll
  for (int j = 0; j < 4; ++j) {
    int c = tid + j * 128;          // chunk id
    rA[j] = c >> 3;                 // row 0..63
    cA[j] = (c & 7) * 8;            // k offset 0..56
    pA[j] = Ab + (size_t)(y_base + rA[j]) * lda + cA[j];
  }
  // B staging bases
  int nB = tid >> 3;                // 0..15
  int kB = (tid & 7) * 8;           // 0..56
  const __bf16* pBt = Bb + (size_t)(x0 + nB) * ldb + kB;  // bTrans: contiguous
  const __bf16* pBn = Bb + x0 + nB;                       // !bTrans: strided rows

  v8f acc = {};
  for (int k0 = 0; k0 < Kd; k0 += KT) {
    if (k0 + KT < Kd)  // pull next A tile toward L2 (global_prefetch_b8)
      __builtin_prefetch(pA[0] + k0 + KT, 0, 1);
#pragma unroll
    for (int j = 0; j < 4; ++j)
      *(uint4*)&As[rA[j]][cA[j]] = *(const uint4*)(pA[j] + k0);
    if (bTrans) {
      *(uint4*)&Bs[nB][kB] = *(const uint4*)(pBt + k0);
    } else {
#pragma unroll
      for (int r = 0; r < 8; ++r)
        Bs[nB][kB + r] = pBn[(size_t)(k0 + kB + r) * ldb];
    }
    __syncthreads();
#pragma unroll
    for (int kk = 0; kk < KT; kk += 32) {
      Frag16 fa, fb;
      const __bf16* arow = &As[w * 16 + ln][kk + 8 * half];
      const __bf16* brow = &Bs[ln][kk + 8 * half];
      fa.q[0] = *(const uint4*)(arow);
      fa.q[1] = *(const uint4*)(arow + 16);
      fb.q[0] = *(const uint4*)(brow);
      fb.q[1] = *(const uint4*)(brow + 16);
      acc = __builtin_amdgcn_wmma_f32_16x16x32_bf16(false, fa.v, false, fb.v,
                                                    (short)0, acc, false, false);
    }
    __syncthreads();
  }
#pragma unroll
  for (int r = 0; r < 8; ++r) {  // C layout: VGPR r -> row r + 8*half, col = lane%16
    size_t o = (size_t)(y_base + w * 16 + r + 8 * half) * ldc + (x0 + ln);
    Cb[o] = acc[r] + (accumulate ? Cb[o] : 0.0f);
  }
}

// ---------------- GRU elementwise ----------------
__global__ void upd_kernel(float* __restrict__ u, const float* __restrict__ G) {
  long i = (long)blockIdx.x * blockDim.x + threadIdx.x;
  if (i >= (long)Bc * Kc * Kc) return;
  long ij = i % (Kc * Kc);
  u[i] = 1.f / (1.f + expf(-(u[i] + G[ij])));
}

__global__ void hcap_kernel(const float* __restrict__ tR, const float* __restrict__ G,
                            const float* __restrict__ A0, __bf16* __restrict__ h) {
  long i = (long)blockIdx.x * blockDim.x + threadIdx.x;
  if (i >= (long)Bc * Kc * Kc) return;
  long ij = i % (Kc * Kc);
  float r = 1.f / (1.f + expf(-(tR[i] + G[ij])));
  h[i] = (__bf16)(r * A0[ij]);
}

__global__ void gru_final_kernel(const float* __restrict__ tH, const float* __restrict__ bh,
                                 const float* __restrict__ u, const float* __restrict__ A0,
                                 float* __restrict__ tA) {
  long i = (long)blockIdx.x * blockDim.x + threadIdx.x;
  if (i >= (long)Bc * Kc * Kc) return;
  long ij = i % (Kc * Kc);
  float h2 = tanhf(tH[i] + bh[ij]);
  float uu = u[i];
  tA[i] += (1.f - uu) * A0[ij] + uu * h2;  // evolved == new_Q numerically
}

// ---------------- Ahat gather + symmetric normalize ----------------
__global__ void di_kernel(const float* __restrict__ Ahat, const int* __restrict__ idx,
                          float* __restrict__ di) {
  int i = blockIdx.x * blockDim.x + threadIdx.x;
  if (i >= Bc * Kc) return;
  int b = i / Kc, j = i % Kc;
  const int* ib = idx + (size_t)b * Kc;
  const float* Ab = Ahat + (size_t)b * Nc * Nc;
  int cj = ib[j];
  float s = 0.f;
  for (int r = 0; r < Kc; ++r) s += Ab[(size_t)ib[r] * Nc + cj];
  di[i] = rsqrtf(s);
}

__global__ void tilde_init_kernel(const float* __restrict__ Ahat, const int* __restrict__ idx,
                                  const float* __restrict__ di, float* __restrict__ tA) {
  long i = (long)blockIdx.x * blockDim.x + threadIdx.x;
  if (i >= (long)Bc * Kc * Kc) return;
  int b = (int)(i / (Kc * Kc));
  int r = (int)((i / Kc) % Kc);
  int c = (int)(i % Kc);
  const int* ib = idx + (size_t)b * Kc;
  float a = Ahat[((size_t)b * Nc + ib[r]) * Nc + ib[c]];
  tA[i] = di[(size_t)b * Kc + c] * a * di[(size_t)b * Kc + r];
}

// ---------------- GCN rank-1 tail ----------------
__global__ void s_kernel(const float* __restrict__ ne, const int* __restrict__ idx,
                         const float* __restrict__ hm, float* __restrict__ s) {
  int gw = (blockIdx.x * blockDim.x + threadIdx.x) >> 5;
  int lane = threadIdx.x & 31;
  if (gw >= Bc * Kc) return;
  int b = gw / Kc, k = gw % Kc;
  int node = idx[(size_t)b * Kc + k];
  const float* row = ne + ((size_t)b * Nc + node) * Dc;
  const float* h = hm + (size_t)b * Dc;
  float acc = 0.f;
  for (int d = lane; d < Dc; d += 32) acc += row[d] * h[d];
  acc = wave_sum(acc);
  if (lane == 0) s[gw] = acc;
}

__global__ void t_kernel(const float* __restrict__ tA, const float* __restrict__ s,
                         float* __restrict__ t) {
  int gw = (blockIdx.x * blockDim.x + threadIdx.x) >> 5;
  int lane = threadIdx.x & 31;
  if (gw >= Bc * Kc) return;
  int b = gw / Kc, r = gw % Kc;
  const float* row = tA + ((size_t)b * Kc + r) * Kc;
  const float* sb = s + (size_t)b * Kc;
  float acc = 0.f;
  for (int j = lane; j < Kc; j += 32) acc += row[j] * sb[j];
  acc = wave_sum(acc);
  if (lane == 0) t[gw] = acc;
}

__global__ void out_kernel(const float* __restrict__ t, const float* __restrict__ g,
                           float* __restrict__ out) {
  long i = (long)blockIdx.x * blockDim.x + threadIdx.x;
  if (i >= (long)Bc * Kc * Dc) return;
  long bk = i / Dc;
  int  e  = (int)(i % Dc);
  float v = t[bk] * g[e];
  out[i] = v > 0.f ? v : 0.f;
}

// ---------------- host launcher ----------------
extern "C" void kernel_launch(void* const* d_in, const int* in_sizes, int n_in,
                              void* d_out, int out_size, void* d_ws, size_t ws_size,
                              hipStream_t stream) {
  (void)in_sizes; (void)n_in; (void)out_size; (void)ws_size;
  const float* Ahat      = (const float*)d_in[0];
  const float* node_embs = (const float*)d_in[1];
  const float* word      = (const float*)d_in[2];
  /* d_in[3] = mask (unused) */
  const float* ht        = (const float*)d_in[4];
  const float* mapper_W  = (const float*)d_in[5];
  const float* mapper_b  = (const float*)d_in[6];
  const float* init_W    = (const float*)d_in[7];
  const float* init_b    = (const float*)d_in[8];
  const float* gmap      = (const float*)d_in[9];
  const float* evA       = (const float*)d_in[10];
  const float* Wu        = (const float*)d_in[11];
  const float* Uu        = (const float*)d_in[12];
  const float* bu        = (const float*)d_in[13];
  const float* Wr        = (const float*)d_in[14];
  const float* Ur        = (const float*)d_in[15];
  const float* br        = (const float*)d_in[16];
  const float* Wh        = (const float*)d_in[17];
  const float* Uh        = (const float*)d_in[18];
  const float* bh        = (const float*)d_in[19];

  float* out_main    = (float*)d_out;                       // [B,K,D]
  float* out_policy  = out_main + (size_t)Bc * Kc * Dc;     // [B]
  float* out_scorer  = out_policy + Bc;                     // [B,D]
  float* out_entropy = out_scorer + (size_t)Bc * Dc;        // [B]

  // bump allocator on workspace
  size_t off = 0;
  char* base = (char*)d_ws;
  auto alloc = [&](size_t bytes) -> void* {
    off = (off + 255) & ~(size_t)255;
    void* p = base + off;
    off += bytes;
    return p;
  };
  const long KK = (long)Kc * Kc;
  float*  hm      = (float*) alloc((size_t)Bc * Dc * 4);
  float*  norm    = (float*) alloc((size_t)Bc * 4);
  float*  scores  = (float*) alloc((size_t)Bc * Nc * 4);
  int*    idx     = (int*)   alloc((size_t)Bc * Kc * 4);
  __bf16* zbf     = (__bf16*)alloc((size_t)Bc * KK * 2);
  __bf16* Wubf    = (__bf16*)alloc((size_t)KK * 2);
  __bf16* Wrbf    = (__bf16*)alloc((size_t)KK * 2);
  __bf16* Whbf    = (__bf16*)alloc((size_t)KK * 2);
  __bf16* Uhbf    = (__bf16*)alloc((size_t)KK * 2);
  float*  Gu      = (float*) alloc((size_t)KK * 4);
  float*  Gr      = (float*) alloc((size_t)KK * 4);
  float*  updB    = (float*) alloc((size_t)Bc * KK * 4);
  float*  tmpR    = (float*) alloc((size_t)Bc * KK * 4);
  __bf16* hcapbf  = (__bf16*)alloc((size_t)Bc * KK * 2);
  float*  tmpH    = (float*) alloc((size_t)Bc * KK * 4);
  __bf16* ngbf    = (__bf16*)alloc((size_t)Bc * Kc * Dc * 2);
  __bf16* wordbf  = (__bf16*)alloc((size_t)Bc * Lc * Dc * 2);
  float*  attn    = (float*) alloc((size_t)Bc * Kc * Lc * 4);
  __bf16* attnbf  = (__bf16*)alloc((size_t)Bc * Kc * Lc * 2);
  float*  tildeA  = (float*) alloc((size_t)Bc * KK * 4);
  float*  di      = (float*) alloc((size_t)Bc * Kc * 4);
  float*  sbuf    = (float*) alloc((size_t)Bc * Kc * 4);
  float*  tbuf    = (float*) alloc((size_t)Bc * Kc * 4);

  dim3 b128(128), b256(256), b512(512);
  const long BKK = (long)Bc * KK;

  // scorer / hm / norms / scores
  linear_tanh_kernel<<<dim3((Bc * Dc) / 256), b256, 0, stream>>>(ht, mapper_W, mapper_b, out_scorer, Dc, Dc);
  linear_tanh_kernel<<<dim3((Bc * Dc) / 256), b256, 0, stream>>>(ht, init_W, init_b, hm, Dc, Dc);
  rownorm_kernel<<<dim3(Bc), b256, 0, stream>>>(out_scorer, norm);
  scores_kernel<<<dim3((Bc * Nc * 32) / 256), b256, 0, stream>>>(node_embs, out_scorer, norm, scores);

  // topk + softmax stats
  topk_kernel<<<dim3(Bc), b512, 0, stream>>>(scores, idx);
  stats_kernel<<<dim3(Bc), b256, 0, stream>>>(scores, idx, out_policy, out_entropy);

  // gathers / conversions for WMMA inputs
  z_gather_kernel<<<dim3((unsigned)(BKK / 256)), b256, 0, stream>>>(node_embs, scores, idx, zbf);
  nodeg_gather_kernel<<<dim3((unsigned)(((long)Bc * Kc * Dc) / 256)), b256, 0, stream>>>(node_embs, idx, ngbf);
  cvt_bf16_kernel<<<dim3((unsigned)(((long)Bc * Lc * Dc) / 256)), b256, 0, stream>>>(word, wordbf, (long)Bc * Lc * Dc);
  cvt_bf16_kernel<<<dim3(KK / 256), b256, 0, stream>>>(Wu, Wubf, KK);
  cvt_bf16_kernel<<<dim3(KK / 256), b256, 0, stream>>>(Wr, Wrbf, KK);
  cvt_bf16_kernel<<<dim3(KK / 256), b256, 0, stream>>>(Wh, Whbf, KK);
  cvt_bf16_kernel<<<dim3(KK / 256), b256, 0, stream>>>(Uh, Uhbf, KK);

  // shared gate terms: Gu = Uu@A + bu, Gr = Ur@A + br
  mmK_bias_kernel<<<dim3(KK / 256), b256, 0, stream>>>(Uu, evA, bu, Gu);
  mmK_bias_kernel<<<dim3(KK / 256), b256, 0, stream>>>(Ur, evA, br, Gr);

  // mat-GRU batched GEMMs (WMMA)
  dim3 gz(Kc / 16, Kc / 64, Bc);
  gemm_bf16_wmma<<<gz, b128, 0, stream>>>(Wubf, 0, Kc, zbf, KK, Kc, 0, updB, KK, Kc, Kc, 0);
  gemm_bf16_wmma<<<gz, b128, 0, stream>>>(Wrbf, 0, Kc, zbf, KK, Kc, 0, tmpR, KK, Kc, Kc, 0);
  upd_kernel<<<dim3((unsigned)(BKK / 256)), b256, 0, stream>>>(updB, Gu);
  hcap_kernel<<<dim3((unsigned)(BKK / 256)), b256, 0, stream>>>(tmpR, Gr, evA, hcapbf);
  gemm_bf16_wmma<<<gz, b128, 0, stream>>>(Whbf, 0, Kc, zbf, KK, Kc, 0, tmpH, KK, Kc, Kc, 0);
  gemm_bf16_wmma<<<gz, b128, 0, stream>>>(Uhbf, 0, Kc, hcapbf, KK, Kc, 0, tmpH, KK, Kc, Kc, 1);

  // Ahat gather + degree normalize -> tildeA
  di_kernel<<<dim3((Bc * Kc) / 256), b256, 0, stream>>>(Ahat, idx, di);
  tilde_init_kernel<<<dim3((unsigned)(BKK / 256)), b256, 0, stream>>>(Ahat, idx, di, tildeA);

  // attention adjacency restricted to topk rows/cols (WMMA)
  dim3 ga(Lc / 16, Kc / 64, Bc);
  gemm_bf16_wmma<<<ga, b128, 0, stream>>>(ngbf, (long)Kc * Dc, Dc, wordbf, (long)Lc * Dc, Dc, 1,
                                          attn, (long)Kc * Lc, Lc, Dc, 0);
  cvt_bf16_kernel<<<dim3((unsigned)(((long)Bc * Kc * Lc) / 256)), b256, 0, stream>>>(attn, attnbf, (long)Bc * Kc * Lc);
  gemm_bf16_wmma<<<gz, b128, 0, stream>>>(attnbf, (long)Kc * Lc, Lc, attnbf, (long)Kc * Lc, Lc, 1,
                                          tildeA, KK, Kc, Lc, 1);

  // tildeA += (1-u)*A + u*tanh(Wh z + Uh hcap + bh)
  gru_final_kernel<<<dim3((unsigned)(BKK / 256)), b256, 0, stream>>>(tmpH, bh, updB, evA, tildeA);

  // GCN rank-1 tail: s = node_g . hm ; t = tildeA @ s ; out = relu(t x gmap)
  s_kernel<<<dim3((Bc * Kc * 32) / 256), b256, 0, stream>>>(node_embs, idx, hm, sbuf);
  t_kernel<<<dim3((Bc * Kc * 32) / 256), b256, 0, stream>>>(tildeA, sbuf, tbuf);
  out_kernel<<<dim3((unsigned)(((long)Bc * Kc * Dc) / 256)), b256, 0, stream>>>(tbuf, gmap, out_main);
}
